// ActionMamba_73186242724492
// MI455X (gfx1250) — matrix operand order repaired and checked by reading, not compiled, not run on hardware
//
#include <hip/hip_runtime.h>
#include <hip/hip_fp16.h>
#include <cstdint>

#define EPS 1e-5f

typedef __attribute__((ext_vector_type(16))) _Float16 v16h;
typedef __attribute__((ext_vector_type(8)))  float    v8f;
typedef __attribute__((ext_vector_type(4)))  float    v4f;
typedef __attribute__((ext_vector_type(4)))  unsigned int u32x4;
typedef __attribute__((ext_vector_type(8)))  int      i32x8;
typedef __attribute__((ext_vector_type(4)))  int      i32x4;

union Frag { v16h v; v4f f4[2]; };

#if __has_builtin(__builtin_amdgcn_tensor_load_to_lds)
#define USE_TDM 1
#else
#define USE_TDM 0
#endif

// Issue a 1-D TDM copy of `bytes` (multiple of 8) from global `ga` into LDS at
// byte-offset `lds` (low 32 bits of a generic LDS pointer). Caller: one wave
// issues, waits tensorcnt, then the block barriers.
__device__ __forceinline__ void tdm_copy_1d(unsigned long long ga, unsigned lds, unsigned bytes)
{
#if USE_TDM
  const unsigned nq = bytes >> 3;                       // 8-byte elements
  u32x4 g0;
  g0[0] = 1u;                                           // count=1, user mode
  g0[1] = lds;                                          // lds_addr
  g0[2] = (unsigned)ga;                                 // global_addr[31:0]
  g0[3] = (unsigned)((ga >> 32) & 0x1ffffffull) | 0x80000000u;  // addr[56:32] | type=2
  i32x8 g1;
  g1[0] = (3 << 16);                                    // data_size=8B, no multicast
  g1[1] = (int)((nq & 0xffffu) << 16);                  // tensor_dim0[15:0]
  g1[2] = (int)(((nq >> 16) & 0xffffu) | (1u << 16));   // tensor_dim0[31:16] | tensor_dim1=1
  g1[3] = (int)((nq & 0xffffu) << 16);                  // tile_dim0 = nq (1-D tile)
  g1[4] = 0;                                            // tile_dim1/2 unused
  g1[5] = (int)nq;                                      // tensor_dim0_stride
  g1[6] = 0; g1[7] = 0;
  i32x4 z4; z4[0] = 0; z4[1] = 0; z4[2] = 0; z4[3] = 0;
  i32x8 z8; z8[0] = 0; z8[1] = 0; z8[2] = 0; z8[3] = 0;
  z8[4] = 0; z8[5] = 0; z8[6] = 0; z8[7] = 0;
  __builtin_amdgcn_tensor_load_to_lds(g0, g1, z4, z4, z8, 0);
  __builtin_amdgcn_s_wait_tensorcnt((short)0);
#else
  (void)ga; (void)lds; (void)bytes;
#endif
}

// ---------------------------------------------------------------------------
// Generic WMMA GEMM:  C[M,N] = A[M,K](f16) * Wt[N,K](f16)^T + bias
//   grid.x = M/128 (8 waves/block, one 16-row A tile per wave)
//   grid.y = N/(16*NT); the NT*16 x K B-panel is TDM-DMA'd into LDS and
//   shared by all 8 waves; each wave reuses its A fragment across NT wmma.
// EPI: 0 = bias, 1 = bias+softplus, 2 = bias+residual(f32)
// ---------------------------------------------------------------------------
template<int EPI, int NT, int KMAX>
__global__ __launch_bounds__(256) void gemm16(
    const _Float16* __restrict__ A, const _Float16* __restrict__ Wt,
    const float* __restrict__ bias, const float* __restrict__ res,
    float* __restrict__ Cf, _Float16* __restrict__ Ch, int N, int K)
{
  __shared__ _Float16 wl[NT * 16 * KMAX];
  const unsigned panelBytes = (unsigned)(NT * 16 * K) * 2u;
#if USE_TDM
  if (threadIdx.x < 32) {
    tdm_copy_1d((unsigned long long)(uintptr_t)(Wt + (size_t)blockIdx.y * NT * 16 * K),
                (unsigned)(uintptr_t)(&wl[0]), panelBytes);
  }
#else
  {
    const v4f* src = (const v4f*)(Wt + (size_t)blockIdx.y * NT * 16 * K);
    v4f* dst = (v4f*)wl;
    const int nv = (int)(panelBytes >> 4);
    for (int i = threadIdx.x; i < nv; i += 256) dst[i] = src[i];
  }
#endif
  __syncthreads();

  const int wave = threadIdx.x >> 5;
  const int lane = threadIdx.x & 31;
  const int lm = lane & 15;
  const int g  = lane >> 4;
  const int mt = blockIdx.x * 8 + wave;
  const size_t arow = (size_t)(mt * 16 + lm) * K;
  __builtin_prefetch(A + arow, 0, 0);

  v8f acc[NT];
  const v8f zero = {0.f, 0.f, 0.f, 0.f, 0.f, 0.f, 0.f, 0.f};
  #pragma unroll
  for (int j = 0; j < NT; ++j) acc[j] = zero;

  for (int kk = 0; kk < K; kk += 32) {
    Frag a;
    a.f4[0] = *(const v4f*)(A + arow + kk + g * 8);
    a.f4[1] = *(const v4f*)(A + arow + kk + 16 + g * 8);
    #pragma unroll
    for (int j = 0; j < NT; ++j) {
      Frag b;
      b.f4[0] = *(const v4f*)(&wl[(j * 16 + lm) * K + kk + g * 8]);
      b.f4[1] = *(const v4f*)(&wl[(j * 16 + lm) * K + kk + 16 + g * 8]);
      acc[j] = __builtin_amdgcn_wmma_f32_16x16x32_f16(false, a.v, false, b.v,
                                                      (short)0, acc[j], false, false);
    }
  }
  #pragma unroll
  for (int j = 0; j < NT; ++j) {
    const int n = (blockIdx.y * NT + j) * 16 + lm;
    const float bv = bias[n];
    #pragma unroll
    for (int r = 0; r < 8; ++r) {
      const size_t m = (size_t)mt * 16 + g * 8 + r;
      float v = acc[j][r] + bv;
      if (EPI == 1) v = (v > 20.f) ? v : log1pf(expf(v));      // softplus
      if (EPI == 2) v += res[m * N + n];                        // residual
      if (Cf) Cf[m * N + n] = v;
      if (Ch) Ch[m * N + n] = (_Float16)v;
    }
  }
}

// ---------------------------------------------------------------------------
// ACE: conv1d(3->64,k=3) + BN + ReLU + channel attention, fully fused.
// One block per (b, j). Output: x_ace f16 (B,T,J,64)
// ---------------------------------------------------------------------------
__global__ __launch_bounds__(256) void ace_kernel(
    const float* __restrict__ x, const float* __restrict__ cw, const float* __restrict__ cb,
    const float* __restrict__ bg, const float* __restrict__ bb,
    const float* __restrict__ bm, const float* __restrict__ bv,
    const float* __restrict__ a1w, const float* __restrict__ a1b,
    const float* __restrict__ a2w, const float* __restrict__ a2b,
    _Float16* __restrict__ xace)
{
  const int T = 1024;
  __shared__ float sx[3 * 1024 + 6];   // zero-padded (t=-1, t=T)
  __shared__ float sw[576];
  __shared__ float S[64], H[64];
  __shared__ float spart[256];
  __shared__ float smean[64];
  __shared__ float sh[16];
  __shared__ float satt[64];
  const int tid = threadIdx.x;
  const int b = blockIdx.x / 21, j = blockIdx.x % 21;

  for (int i = tid; i < 576; i += 256) sw[i] = cw[i];
  if (tid < 64) {
    const float s = bg[tid] * rsqrtf(bv[tid] + EPS);
    S[tid] = s;
    H[tid] = (cb[tid] - bm[tid]) * s + bb[tid];
  }
  if (tid < 6) sx[tid < 3 ? tid : (3 + 3 * T + (tid - 3))] = 0.f;
  for (int i = tid; i < 3 * T; i += 256) {
    const int t = i / 3, c = i - t * 3;
    sx[3 + i] = x[((size_t)b * T + t) * 63 + j * 3 + c];
  }
  __syncthreads();

  const int o = tid & 63, sl = tid >> 6;
  const float* wo = sw + o * 9;
  float psum = 0.f;
  for (int t = sl; t < T; t += 4) {
    const float* xt = sx + 3 * t;     // xt[0..2]=x[t-1], [3..5]=x[t], [6..8]=x[t+1]
    const float z = wo[0]*xt[0] + wo[1]*xt[3] + wo[2]*xt[6]
                  + wo[3]*xt[1] + wo[4]*xt[4] + wo[5]*xt[7]
                  + wo[6]*xt[2] + wo[7]*xt[5] + wo[8]*xt[8];
    psum += fmaxf(z * S[o] + H[o], 0.f);
  }
  spart[tid] = psum;
  __syncthreads();
  if (tid < 64)
    smean[tid] = (spart[tid] + spart[tid + 64] + spart[tid + 128] + spart[tid + 192]) / (float)T;
  __syncthreads();
  if (tid < 16) {
    float s = a1b[tid];
    for (int k = 0; k < 64; ++k) s += smean[k] * a1w[k * 16 + tid];
    sh[tid] = fmaxf(s, 0.f);
  }
  __syncthreads();
  if (tid < 64) {
    float s = a2b[tid];
    for (int k = 0; k < 16; ++k) s += sh[k] * a2w[k * 64 + tid];
    satt[tid] = 1.f / (1.f + expf(-s));
  }
  __syncthreads();
  const float att = satt[o];
  for (int t = sl; t < T; t += 4) {
    const float* xt = sx + 3 * t;
    const float z = wo[0]*xt[0] + wo[1]*xt[3] + wo[2]*xt[6]
                  + wo[3]*xt[1] + wo[4]*xt[4] + wo[5]*xt[7]
                  + wo[6]*xt[2] + wo[7]*xt[5] + wo[8]*xt[8];
    const float v = fmaxf(z * S[o] + H[o], 0.f) * att;
    xace[(((size_t)b * T + t) * 21 + j) * 64 + o] = (_Float16)v;
  }
}

// ---------------------------------------------------------------------------
// Fused GCN: per (b, 8-timestep chunk): einsum(A) -> WMMA 64->128 (BN+ReLU)
// -> einsum(A) -> WMMA 128->128 (BN+ReLU) -> partial channel sums.
// 168 real rows padded to 176 (11 mtiles); pad rows masked out of the sum.
// ---------------------------------------------------------------------------
__global__ __launch_bounds__(256) void gcn_fused(
    const _Float16* __restrict__ xace, const float* __restrict__ Aadj,
    const _Float16* __restrict__ W1t, const float* __restrict__ b1,
    const float* __restrict__ g1, const float* __restrict__ be1,
    const float* __restrict__ m1, const float* __restrict__ v1,
    const _Float16* __restrict__ W2t, const float* __restrict__ b2,
    const float* __restrict__ g2, const float* __restrict__ be2,
    const float* __restrict__ m2, const float* __restrict__ v2,
    float* __restrict__ part)
{
  __shared__ float sA[441];
  __shared__ _Float16 sW1[128 * 64];
  __shared__ _Float16 sW2[128 * 128];
  __shared__ float S1[128], H1[128], S2[128], H2[128];
  __shared__ _Float16 sBuf[176 * 128];   // [x(176x64) | xg(176x64)] then h2(176x128)
  __shared__ _Float16 sG1[176 * 128];
  __shared__ float accC[128];
  const int tid = threadIdx.x;
  const int b = blockIdx.x >> 7;         // 128 chunks of 8 timesteps per batch
  const int tc = blockIdx.x & 127;
  _Float16* sXg = sBuf + 176 * 64;

  for (int i = tid; i < 441; i += 256) sA[i] = Aadj[i];
  if (tid < 128) {
    float s = g1[tid] * rsqrtf(v1[tid] + EPS);
    S1[tid] = s; H1[tid] = b1[tid] * s + be1[tid] - m1[tid] * s;
    s = g2[tid] * rsqrtf(v2[tid] + EPS);
    S2[tid] = s; H2[tid] = b2[tid] * s + be2[tid] - m2[tid] * s;
    accC[tid] = 0.f;
  }
  for (int i = tid; i < (128 * 64) / 8; i += 256)  ((v4f*)sW1)[i] = ((const v4f*)W1t)[i];
  for (int i = tid; i < (128 * 128) / 8; i += 256) ((v4f*)sW2)[i] = ((const v4f*)W2t)[i];
  {
    const v4f* src = (const v4f*)(xace + ((size_t)b * 1024 + tc * 8) * 1344);
    for (int i = tid; i < (168 * 64) / 8; i += 256) ((v4f*)sBuf)[i] = src[i];
  }
  for (int i = tid; i < 512; i += 256) {           // zero pad rows (A operand)
    sBuf[168 * 64 + i] = (_Float16)0.f;
    sXg [168 * 64 + i] = (_Float16)0.f;
  }
  __syncthreads();

  // einsum1: xg[t,u,:] = sum_v A[u,v] * x[t,v,:]
  for (int i = tid; i < 168 * 64; i += 256) {
    const int c = i & 63, row = i >> 6;
    const int u = row % 21, tl = row / 21;
    const _Float16* xb = sBuf + (tl * 21) * 64 + c;
    float s = 0.f;
    #pragma unroll
    for (int vv = 0; vv < 21; ++vv) s += sA[u * 21 + vv] * (float)xb[vv * 64];
    sXg[row * 64 + c] = (_Float16)s;
  }
  __syncthreads();

  const int wave = tid >> 5, lane = tid & 31, lm = lane & 15, g = lane >> 4;
  // GEMM1: (176x64) @ W1(64x128), relu(bn)
  for (int tix = wave; tix < 88; tix += 8) {
    const int mt = tix >> 3, nt = tix & 7;
    v8f acc = {0.f, 0.f, 0.f, 0.f, 0.f, 0.f, 0.f, 0.f};
    #pragma unroll
    for (int kk = 0; kk < 64; kk += 32) {
      Frag a, bb;
      a.f4[0] = *(const v4f*)(sXg + (mt * 16 + lm) * 64 + kk + g * 8);
      a.f4[1] = *(const v4f*)(sXg + (mt * 16 + lm) * 64 + kk + 16 + g * 8);
      bb.f4[0] = *(const v4f*)(sW1 + (nt * 16 + lm) * 64 + kk + g * 8);
      bb.f4[1] = *(const v4f*)(sW1 + (nt * 16 + lm) * 64 + kk + 16 + g * 8);
      acc = __builtin_amdgcn_wmma_f32_16x16x32_f16(false, a.v, false, bb.v,
                                                   (short)0, acc, false, false);
    }
    const int n = nt * 16 + lm;
    #pragma unroll
    for (int r = 0; r < 8; ++r) {
      const int m = mt * 16 + g * 8 + r;
      sG1[m * 128 + n] = (_Float16)fmaxf(acc[r] * S1[n] + H1[n], 0.f);
    }
  }
  __syncthreads();

  // einsum2 into h2 (reuses sBuf)
  for (int i = tid; i < 168 * 128; i += 256) {
    const int c = i & 127, row = i >> 7;
    const int u = row % 21, tl = row / 21;
    const _Float16* gb = sG1 + (tl * 21) * 128 + c;
    float s = 0.f;
    #pragma unroll
    for (int vv = 0; vv < 21; ++vv) s += sA[u * 21 + vv] * (float)gb[vv * 128];
    sBuf[row * 128 + c] = (_Float16)s;
  }
  for (int i = tid; i < 1024; i += 256) sBuf[168 * 128 + i] = (_Float16)0.f;
  __syncthreads();

  // GEMM2: (176x128) @ W2(128x128), relu(bn), accumulate channel sums
  for (int tix = wave; tix < 88; tix += 8) {
    const int mt = tix >> 3, nt = tix & 7;
    v8f acc = {0.f, 0.f, 0.f, 0.f, 0.f, 0.f, 0.f, 0.f};
    #pragma unroll
    for (int kk = 0; kk < 128; kk += 32) {
      Frag a, bb;
      a.f4[0] = *(const v4f*)(sBuf + (mt * 16 + lm) * 128 + kk + g * 8);
      a.f4[1] = *(const v4f*)(sBuf + (mt * 16 + lm) * 128 + kk + 16 + g * 8);
      bb.f4[0] = *(const v4f*)(sW2 + (nt * 16 + lm) * 128 + kk + g * 8);
      bb.f4[1] = *(const v4f*)(sW2 + (nt * 16 + lm) * 128 + kk + 16 + g * 8);
      acc = __builtin_amdgcn_wmma_f32_16x16x32_f16(false, a.v, false, bb.v,
                                                   (short)0, acc, false, false);
    }
    const int n = nt * 16 + lm;
    float p = 0.f;
    #pragma unroll
    for (int r = 0; r < 8; ++r) {
      const int m = mt * 16 + g * 8 + r;
      const float v = fmaxf(acc[r] * S2[n] + H2[n], 0.f);
      p += (m < 168) ? v : 0.f;
    }
    atomicAdd(&accC[n], p);
  }
  __syncthreads();
  if (tid < 128) part[(size_t)blockIdx.x * 128 + tid] = accC[tid];
}

__global__ void gcn_reduce(const float* __restrict__ part, float* __restrict__ spatial)
{
  const int b = blockIdx.x, n = threadIdx.x;   // 128 threads
  float s = 0.f;
  for (int c = 0; c < 128; ++c) s += part[((size_t)b * 128 + c) * 128 + n];
  spatial[b * 128 + n] = s / (1024.f * 21.f);
}

// ---------------------------------------------------------------------------
// Mamba helpers
// ---------------------------------------------------------------------------
__global__ __launch_bounds__(256) void dwconv_silu(
    const _Float16* __restrict__ xp, const float* __restrict__ cw,
    const float* __restrict__ cb, _Float16* __restrict__ xc)
{
  const int T = 1024;
  const size_t i = (size_t)blockIdx.x * 256 + threadIdx.x;   // B*T*512
  const int e = (int)(i & 511);
  const size_t bt = i >> 9;
  const int t = (int)(bt & (size_t)(T - 1));
  const float xm = (t > 0)     ? (float)xp[(bt - 1) * 1024 + e] : 0.f;
  const float x0 =               (float)xp[bt * 1024 + e];
  const float xq = (t < T - 1) ? (float)xp[(bt + 1) * 1024 + e] : 0.f;
  const float z = cw[e * 3] * xm + cw[e * 3 + 1] * x0 + cw[e * 3 + 2] * xq + cb[e];
  xc[bt * 512 + e] = (_Float16)(z / (1.f + expf(-z)));       // silu
}

__global__ __launch_bounds__(256) void abd_kernel(
    const _Float16* __restrict__ dt, const float* __restrict__ Bt,
    const float* __restrict__ Dt, float* __restrict__ a, float* __restrict__ bd)
{
  const int lane = threadIdx.x & 31;
  const size_t row = ((size_t)blockIdx.x * 256 + threadIdx.x) >> 5;
  float s = 0.f;
  for (int k = lane; k < 512; k += 32) s += (float)dt[row * 512 + k];
  #pragma unroll
  for (int off = 16; off > 0; off >>= 1) s += __shfl_down(s, off);
  if (lane == 0) {
    const float aa = fminf(10.f, fmaxf(-10.f, -s / 512.f));
    a[row] = expf(aa);
  }
  if (lane < 16) bd[row * 16 + lane] = Bt[row * 16 + lane] * Dt[row * 16 + lane];
}

__global__ void scan_kernel(const float* __restrict__ a, const float* __restrict__ bd,
                            const float* __restrict__ Ct, float* __restrict__ ys)
{
  const int T = 1024;
  const int b = blockIdx.x, lane = threadIdx.x;   // 32 threads
  float st = 0.f;
  for (int t = 0; t < T; ++t) {
    const size_t row = (size_t)b * T + t;
    const float at = a[row];
    float p = 0.f;
    if (lane < 16) {
      st = st * at + bd[row * 16 + lane];
      p = Ct[row * 16 + lane] * st;
    }
    #pragma unroll
    for (int off = 16; off > 0; off >>= 1) p += __shfl_down(p, off);
    if (lane == 0) ys[row] = p;
  }
}

__global__ __launch_bounds__(256) void gate_kernel(
    const _Float16* __restrict__ xp, const float* __restrict__ ys, _Float16* __restrict__ y)
{
  const size_t i = (size_t)blockIdx.x * 256 + threadIdx.x;
  const int e = (int)(i & 511);
  const size_t bt = i >> 9;
  const float gv = (float)xp[bt * 1024 + 512 + e];
  y[bt * 512 + e] = (_Float16)(ys[bt] * (gv / (1.f + expf(-gv))));
}

// ---------------------------------------------------------------------------
// Tail: LN + mean over T, fusion head
// ---------------------------------------------------------------------------
__device__ __forceinline__ float blockSum256(float v, float* red)
{
  const int n = threadIdx.x;
  red[n] = v; __syncthreads();
  #pragma unroll
  for (int s = 128; s > 0; s >>= 1) { if (n < s) red[n] += red[n + s]; __syncthreads(); }
  const float r = red[0]; __syncthreads();
  return r;
}

__global__ __launch_bounds__(256) void lnmean_kernel(
    const float* __restrict__ t, const float* __restrict__ g,
    const float* __restrict__ be, float* __restrict__ temporal)
{
  const int T = 1024;
  __shared__ float red[256];
  const int b = blockIdx.x, n = threadIdx.x;
  float acc = 0.f;
  for (int tt = 0; tt < T; ++tt) {
    const float x = t[((size_t)b * T + tt) * 256 + n];
    const float m = blockSum256(x, red) * (1.f / 256.f);
    const float d = x - m;
    const float var = blockSum256(d * d, red) * (1.f / 256.f);
    acc += d * rsqrtf(var + EPS) * g[n] + be[n];
  }
  temporal[b * 256 + n] = acc / (float)T;
}

__device__ __forceinline__ float geluf(float x)
{ return 0.5f * x * (1.f + erff(x * 0.70710678118654752f)); }

__global__ __launch_bounds__(256) void fusion_kernel(
    const float* __restrict__ spatial, const float* __restrict__ temporal,
    const float* __restrict__ w1, const float* __restrict__ b1,
    const float* __restrict__ l1g, const float* __restrict__ l1b,
    const float* __restrict__ w2, const float* __restrict__ b2,
    const float* __restrict__ l2g, const float* __restrict__ l2b,
    const float* __restrict__ w3, const float* __restrict__ b3,
    float* __restrict__ out)
{
  __shared__ float f[384], h1[256], h2[128], red[256];
  const int n = threadIdx.x;
  for (int b = 0; b < 32; ++b) {
    for (int i = n; i < 384; i += 256)
      f[i] = (i < 128) ? spatial[b * 128 + i] : temporal[b * 256 + (i - 128)];
    __syncthreads();
    float v = b1[n];
    for (int k = 0; k < 384; ++k) v += f[k] * w1[k * 256 + n];
    const float m = blockSum256(v, red) * (1.f / 256.f);
    const float d = v - m;
    const float var = blockSum256(d * d, red) * (1.f / 256.f);
    h1[n] = geluf(d * rsqrtf(var + EPS) * l1g[n] + l1b[n]);
    __syncthreads();
    float v2 = 0.f;
    if (n < 128) { v2 = b2[n]; for (int k = 0; k < 256; ++k) v2 += h1[k] * w2[k * 128 + n]; }
    const float m2 = blockSum256((n < 128) ? v2 : 0.f, red) * (1.f / 128.f);
    const float d2 = (n < 128) ? (v2 - m2) : 0.f;
    const float var2 = blockSum256(d2 * d2, red) * (1.f / 128.f);
    if (n < 128) h2[n] = geluf(d2 * rsqrtf(var2 + EPS) * l2g[n] + l2b[n]);
    __syncthreads();
    if (n < 4) {
      float o = b3[n];
      for (int k = 0; k < 128; ++k) o += h2[k] * w3[k * 4 + n];
      out[b * 4 + n] = o;
    }
    __syncthreads();
  }
}

// fp32 (K,N) -> fp16 (N,K) weight pack for B-fragment-friendly loads
__global__ void packWT(const float* __restrict__ w, _Float16* __restrict__ o, int K, int N)
{
  const int i = blockIdx.x * 256 + threadIdx.x;
  if (i >= K * N) return;
  const int nn = i / K, kk = i - nn * K;
  o[i] = (_Float16)w[(size_t)kk * N + nn];
}

// ---------------------------------------------------------------------------
extern "C" void kernel_launch(void* const* d_in, const int* in_sizes, int n_in,
                              void* d_out, int out_size, void* d_ws, size_t ws_size,
                              hipStream_t stream)
{
  (void)in_sizes; (void)n_in; (void)out_size; (void)ws_size;
  const int B = 32, T = 1024;
  const size_t BT = (size_t)B * T;
  auto F = [&](int i) { return (const float*)d_in[i]; };

  char* ws = (char*)d_ws;
  size_t off = 0;
  auto alloc = [&](size_t bytes) { size_t r = off; off += (bytes + 255) & ~(size_t)255; return r; };

  const size_t oXACE = alloc(BT * 21 * 64 * 2);          // 84MB; reused as XP (67MB)
  _Float16* XACE = (_Float16*)(ws + oXACE);
  _Float16* XP   = XACE;
  float*    TF   = (float*)(ws + alloc(BT * 256 * 4));
  _Float16* TH   = (_Float16*)(ws + alloc(BT * 256 * 2));
  _Float16* XC   = (_Float16*)(ws + alloc(BT * 512 * 2));   // conv out, later gated y
  _Float16* DTH  = (_Float16*)(ws + alloc(BT * 512 * 2));
  float* BO  = (float*)(ws + alloc(BT * 16 * 4));
  float* CO  = (float*)(ws + alloc(BT * 16 * 4));
  float* DO_ = (float*)(ws + alloc(BT * 16 * 4));
  float* AA  = (float*)(ws + alloc(BT * 4));
  float* BD  = (float*)(ws + alloc(BT * 16 * 4));
  float* YS  = (float*)(ws + alloc(BT * 4));
  float* GP  = (float*)(ws + alloc((size_t)B * 128 * 128 * 4));
  float* SPAT = (float*)(ws + alloc(B * 128 * 4));
  float* TEMP = (float*)(ws + alloc(B * 256 * 4));
  _Float16* PTMIN = (_Float16*)(ws + alloc(1344 * 256 * 2));
  _Float16* PG1   = (_Float16*)(ws + alloc(64 * 128 * 2));
  _Float16* PG2   = (_Float16*)(ws + alloc(128 * 128 * 2));
  _Float16 *PIN[4], *PDT[4], *PB[4], *PC[4], *PD[4], *PO[4];
  for (int l = 0; l < 4; ++l) {
    PIN[l] = (_Float16*)(ws + alloc(256 * 1024 * 2));
    PDT[l] = (_Float16*)(ws + alloc(512 * 512 * 2));
    PB[l]  = (_Float16*)(ws + alloc(512 * 16 * 2));
    PC[l]  = (_Float16*)(ws + alloc(512 * 16 * 2));
    PD[l]  = (_Float16*)(ws + alloc(512 * 16 * 2));
    PO[l]  = (_Float16*)(ws + alloc(512 * 256 * 2));
  }

  auto pack = [&](const float* w, _Float16* dst, int K, int N) {
    packWT<<<(K * N + 255) / 256, 256, 0, stream>>>(w, dst, K, N);
  };
  pack(F(24), PTMIN, 1344, 256);
  pack(F(12), PG1, 64, 128);
  pack(F(18), PG2, 128, 128);
  for (int l = 0; l < 4; ++l) {
    const int p = 28 + l * 14;
    pack(F(p + 0),  PIN[l], 256, 1024);
    pack(F(p + 4),  PDT[l], 512, 512);
    pack(F(p + 6),  PB[l],  512, 16);
    pack(F(p + 8),  PC[l],  512, 16);
    pack(F(p + 10), PD[l],  512, 16);
    pack(F(p + 12), PO[l],  512, 256);
  }

  // ACE -> x_ace f16 (B,T,J,64)
  ace_kernel<<<B * 21, 256, 0, stream>>>(F(0), F(2), F(3), F(4), F(5), F(6), F(7),
                                         F(8), F(9), F(10), F(11), XACE);
  // GCN path (fused, chunked) -> spatial (B,128)
  gcn_fused<<<B * 128, 256, 0, stream>>>(XACE, F(1), PG1, F(13), F(14), F(15), F(16), F(17),
                                         PG2, F(19), F(20), F(21), F(22), F(23), GP);
  gcn_reduce<<<B, 128, 0, stream>>>(GP, SPAT);

  // tm_in: (BT,1344)@(1344,256) -> t (f32 + f16 shadow)
  gemm16<0, 1, 1344><<<dim3(256, 16), 256, 0, stream>>>(XACE, PTMIN, F(25), nullptr,
                                                        TF, TH, 256, 1344);

  for (int l = 0; l < 4; ++l) {
    const int p = 28 + l * 14;
    gemm16<0, 2, 256><<<dim3(256, 32), 256, 0, stream>>>(TH, PIN[l], F(p + 1), nullptr,
                                                         nullptr, XP, 1024, 256);
    dwconv_silu<<<(int)(BT * 512 / 256), 256, 0, stream>>>(XP, F(p + 2), F(p + 3), XC);
    gemm16<1, 2, 512><<<dim3(256, 16), 256, 0, stream>>>(XC, PDT[l], F(p + 5), nullptr,
                                                         nullptr, DTH, 512, 512);
    gemm16<0, 1, 512><<<dim3(256, 1), 256, 0, stream>>>(XC, PB[l], F(p + 7),  nullptr,
                                                        BO,  nullptr, 16, 512);
    gemm16<0, 1, 512><<<dim3(256, 1), 256, 0, stream>>>(XC, PC[l], F(p + 9),  nullptr,
                                                        CO,  nullptr, 16, 512);
    gemm16<0, 1, 512><<<dim3(256, 1), 256, 0, stream>>>(XC, PD[l], F(p + 11), nullptr,
                                                        DO_, nullptr, 16, 512);
    abd_kernel<<<(int)(BT / 8), 256, 0, stream>>>(DTH, BO, DO_, AA, BD);
    scan_kernel<<<B, 32, 0, stream>>>(AA, BD, CO, YS);
    gate_kernel<<<(int)(BT * 512 / 256), 256, 0, stream>>>(XP, YS, XC);
    gemm16<2, 2, 512><<<dim3(256, 8), 256, 0, stream>>>(XC, PO[l], F(p + 13), TF,
                                                        TF, TH, 256, 512);
  }

  lnmean_kernel<<<B, 256, 0, stream>>>(TF, F(26), F(27), TEMP);
  fusion_kernel<<<1, 256, 0, stream>>>(SPAT, TEMP, F(84), F(85), F(86), F(87),
                                       F(88), F(89), F(90), F(91), F(92), F(93),
                                       (float*)d_out);
}
